// GTConvBlock_84559316123820
// MI455X (gfx1250) — matrix operand order repaired and checked by reading, not compiled, not run on hardware
//
#include <hip/hip_runtime.h>
#include <hip/hip_bf16.h>

typedef __attribute__((ext_vector_type(16))) __bf16 v16bf;
typedef __attribute__((ext_vector_type(8)))  __bf16 v8bf;
typedef __attribute__((ext_vector_type(8)))  float  v8f;

#define B_    8
#define C_    64
#define CH_   32
#define T_    512
#define F_    128
#define HID_  64
#define GH_   64
#define TILE_T 2
#define NSLICE (TILE_T + 2)
#define EPS_LN 1e-6f
#define EPS_BN 1e-5f

// A fragment: lane holds row m; elements 0..15 are K = khalf*16+e (contiguous,
// matching B).  Both halves are contiguous 16B runs in a K-major row -> 2x ds_load_b128.
__device__ inline v16bf load_afrag(const __bf16* row, int r0, int khalf) {
    v8bf lo = *(const v8bf*)(row + r0 + khalf * 16);
    v8bf hi = *(const v8bf*)(row + r0 + khalf * 16 + 8);
    return __builtin_shufflevector(lo, hi, 0, 1, 2, 3, 4, 5, 6, 7,
                                           8, 9, 10, 11, 12, 13, 14, 15);
}

// ---------------------------------------------------------------------------
// Phase 1: fused pw1 -> LN1 -> PReLU -> depthwise 3x3 (causal T) -> BN ->
// PReLU -> pw2 -> LN2.  One block per (b, t-tile).  GEMMs via WMMA bf16 with
// K-contiguous B staging (im2col transposed in LDS).
// ---------------------------------------------------------------------------
__launch_bounds__(256, 1)
__global__ void fused_conv_kernel(
    const float* __restrict__ x,
    const float* __restrict__ pw1_w, const float* __restrict__ pw1_b,
    const float* __restrict__ ln1_g, const float* __restrict__ ln1_b,
    const float* __restrict__ prelu1,
    const float* __restrict__ dw_w,  const float* __restrict__ dw_b,
    const float* __restrict__ bn_g,  const float* __restrict__ bn_b,
    const float* __restrict__ bn_mean, const float* __restrict__ bn_var,
    const float* __restrict__ prelu2,
    const float* __restrict__ pw2_w, const float* __restrict__ pw2_b,
    const float* __restrict__ ln2_g, const float* __restrict__ ln2_b,
    float* __restrict__ h_ws, float* __restrict__ zt_ws)
{
    __shared__ __attribute__((aligned(32))) __bf16 w1s[HID_][96];   // 12 KB
    __shared__ __attribute__((aligned(32))) __bf16 w2s[CH_][HID_];  // 4 KB
    __shared__ __attribute__((aligned(32))) __bf16 bts[F_][96];     // 24 KB im2col^T
    __shared__ __bf16 as_[NSLICE][HID_][F_ + 2];                    // 65 KB
    __shared__ __attribute__((aligned(32))) __bf16 psT[F_][HID_];   // 16 KB
    __shared__ float  h2s[CH_][F_];                                 // 16 KB
    __shared__ float  par[4 * HID_ + 2 * CH_ + 64];                 // affine params

    float* pw1b_s = par;                 // 64
    float* g1s    = par + 64;            // 64
    float* b1s    = par + 128;           // 64
    float* pw2b_s = par + 192;           // 32
    float* g2s    = par + 224;           // 32
    float* b2s    = par + 256;           // 32

    const int tid  = threadIdx.x;
    const int lane = tid & 31;
    const int wv   = tid >> 5;                  // wave id 0..7 -> N-tile
    const int b    = blockIdx.x / (T_ / TILE_T);
    const int t0   = (blockIdx.x % (T_ / TILE_T)) * TILE_T;

    // weights / params -> LDS
    for (int i = tid; i < HID_ * 96; i += 256)
        ((__bf16*)w1s)[i] = (__bf16)pw1_w[i];
    for (int i = tid; i < CH_ * HID_; i += 256)
        ((__bf16*)w2s)[i] = (__bf16)pw2_w[i];
    if (tid < HID_) {
        pw1b_s[tid] = pw1_b[tid];
        g1s[tid]    = ln1_g[tid];
        b1s[tid]    = ln1_b[tid];
    } else if (tid < HID_ + CH_) {
        int c = tid - HID_;
        pw2b_s[c] = pw2_b[c];
        g2s[c]    = ln2_g[c];
        b2s[c]    = ln2_b[c];
    }
    for (int i = tid; i < NSLICE * HID_; i += 256) {
        int s = i / HID_, c = i % HID_;
        as_[s][c][0] = (__bf16)0.f; as_[s][c][F_ + 1] = (__bf16)0.f;
    }

    const float a1 = prelu1[0];
    const float a2 = prelu2[0];
    const int   mrow  = lane & 15;   // row within 16-tile (A) / col (B)
    const int   khalf = lane >> 4;   // K-half select per ISA layout
    const int   f0    = wv * 16;     // this wave's N-tile columns
    __syncthreads();

    // ---------------- Stage A: pw1 GEMM (64x96 @ 96x128) + LN1 + PReLU -----
    for (int s = 0; s < NSLICE; ++s) {
        const int t = t0 - 2 + s;
        if (t >= 0) {
            // Build bts[f][3c+kk] = x1[c][f+kk-1]; coalesced global reads.
            for (int i = tid; i < CH_ * F_; i += 256) {
                int c = i >> 7, fq = i & 127;
                __bf16 v = (__bf16)x[((b * C_ + c) * T_ + t) * F_ + fq];
                if (fq < F_ - 1) bts[fq + 1][3 * c + 0] = v;
                bts[fq][3 * c + 1] = v;
                if (fq > 0) bts[fq - 1][3 * c + 2] = v;
            }
            if (tid < CH_) {   // F-edge zero pad
                bts[0][3 * tid + 0] = (__bf16)0.f;
                bts[F_ - 1][3 * tid + 2] = (__bf16)0.f;
            }
        }
        __syncthreads();

        if (t < 0) {  // reference zero-pads AFTER stage A
            for (int i = tid; i < HID_ * F_; i += 256)
                as_[s][i / F_][1 + (i % F_)] = (__bf16)0.f;
            __syncthreads();
            continue;
        }

        v8f acc[4];
        #pragma unroll
        for (int mt = 0; mt < 4; ++mt)
            #pragma unroll
            for (int i = 0; i < 8; ++i) acc[mt][i] = 0.f;

        #pragma unroll
        for (int ks = 0; ks < 3; ++ks) {
            const int r0 = ks * 32;
            v16bf bf = *(const v16bf*)(&bts[f0 + mrow][r0 + khalf * 16]);
            #pragma unroll
            for (int mt = 0; mt < 4; ++mt) {
                v16bf af = load_afrag(&w1s[mt * 16 + mrow][0], r0, khalf);
                acc[mt] = __builtin_amdgcn_wmma_f32_16x16x32_bf16(
                    false, af, false, bf, (short)0, acc[mt], false, false);
            }
        }
        // LN over 64 channels of column f0+mrow (lane pair l / l^16)
        float s1 = 0.f, s2 = 0.f;
        float vals[4][8];
        #pragma unroll
        for (int mt = 0; mt < 4; ++mt)
            #pragma unroll
            for (int i = 0; i < 8; ++i) {
                int ch = mt * 16 + khalf * 8 + i;
                float v = acc[mt][i] + pw1b_s[ch];
                vals[mt][i] = v; s1 += v; s2 += v * v;
            }
        s1 += __shfl_xor(s1, 16);
        s2 += __shfl_xor(s2, 16);
        float mu   = s1 * (1.f / 64.f);
        float var  = s2 * (1.f / 64.f) - mu * mu;
        float rstd = rsqrtf(var + EPS_LN);
        #pragma unroll
        for (int mt = 0; mt < 4; ++mt)
            #pragma unroll
            for (int i = 0; i < 8; ++i) {
                int ch = mt * 16 + khalf * 8 + i;
                float v = (vals[mt][i] - mu) * rstd * g1s[ch] + b1s[ch];
                v = (v >= 0.f) ? v : a1 * v;
                as_[s][ch][1 + f0 + mrow] = (__bf16)v;
            }
        __syncthreads();   // before next slice rebuilds bts
    }

    // ---------------- Stage B: dw conv + BN + PReLU + pw2 + LN2 ------------
    const int dch   = tid >> 2;            // depthwise: thread -> (ch, 32 f)
    const int fbase = (tid & 3) * 32;
    float dwk[9];
    #pragma unroll
    for (int j = 0; j < 9; ++j) dwk[j] = dw_w[dch * 9 + j];
    const float bnin = bn_g[dch] * rsqrtf(bn_var[dch] + EPS_BN);
    const float bnsh = bn_b[dch] - bn_mean[dch] * bnin;
    const float dwb  = dw_b[dch];

    for (int so = 0; so < TILE_T; ++so) {
        const int t = t0 + so;
        for (int ff = 0; ff < 32; ++ff) {   // depthwise 3x3 -> psT (transposed)
            int f = fbase + ff;
            float a = dwb;
            #pragma unroll
            for (int kt = 0; kt < 3; ++kt)
                #pragma unroll
                for (int kf = 0; kf < 3; ++kf)
                    a += dwk[kt * 3 + kf] * (float)as_[so + kt][dch][f + kf];
            float v = a * bnin + bnsh;
            v = (v >= 0.f) ? v : a2 * v;
            psT[f][dch] = (__bf16)v;
        }
        __syncthreads();

        // pw2 GEMM (32x64 @ 64x128) + LN2 over 32 channels
        v8f acc2[2];
        #pragma unroll
        for (int mt = 0; mt < 2; ++mt)
            #pragma unroll
            for (int i = 0; i < 8; ++i) acc2[mt][i] = 0.f;

        #pragma unroll
        for (int ks = 0; ks < 2; ++ks) {
            v16bf bf = *(const v16bf*)(&psT[f0 + mrow][ks * 32 + khalf * 16]);
            #pragma unroll
            for (int mt = 0; mt < 2; ++mt) {
                v16bf af = load_afrag(&w2s[mt * 16 + mrow][0], ks * 32, khalf);
                acc2[mt] = __builtin_amdgcn_wmma_f32_16x16x32_bf16(
                    false, af, false, bf, (short)0, acc2[mt], false, false);
            }
        }
        float s1 = 0.f, s2 = 0.f;
        float vals[2][8];
        #pragma unroll
        for (int mt = 0; mt < 2; ++mt)
            #pragma unroll
            for (int i = 0; i < 8; ++i) {
                int ch = mt * 16 + khalf * 8 + i;
                float v = acc2[mt][i] + pw2b_s[ch];
                vals[mt][i] = v; s1 += v; s2 += v * v;
            }
        s1 += __shfl_xor(s1, 16);
        s2 += __shfl_xor(s2, 16);
        float mu   = s1 * (1.f / 32.f);
        float var  = s2 * (1.f / 32.f) - mu * mu;
        float rstd = rsqrtf(var + EPS_LN);
        #pragma unroll
        for (int mt = 0; mt < 2; ++mt)
            #pragma unroll
            for (int i = 0; i < 8; ++i) {
                int ch = mt * 16 + khalf * 8 + i;
                h2s[ch][f0 + mrow] =
                    (vals[mt][i] - mu) * rstd * g2s[ch] + b2s[ch];
            }
        __syncthreads();

        // coalesced float4 writeback of h, plus zt = mean_F h^2
        for (int i = tid; i < CH_ * (F_ / 4); i += 256) {
            int c = i >> 5, f4 = i & 31;
            float4 v = ((const float4*)h2s)[(c * F_ >> 2) + f4];
            ((float4*)h_ws)[((((b * CH_ + c) * T_ + t) * F_) >> 2) + f4] = v;
        }
        if (tid < CH_) {
            float sum = 0.f;
            for (int f = 0; f < F_; ++f) { float v = h2s[tid][f]; sum += v * v; }
            zt_ws[(b * T_ + t) * CH_ + tid] = sum * (1.f / F_);
        }
        __syncthreads();   // before next so overwrites psT / h2s
    }
}

// ---------------------------------------------------------------------------
// Phase 2a: xg[b,t,g] = wih[g,:] . zt[b,t,:] + bih[g]
// ---------------------------------------------------------------------------
__global__ void xg_kernel(const float* __restrict__ zt,
                          const float* __restrict__ wih,
                          const float* __restrict__ bih,
                          float* __restrict__ xg)
{
    __shared__ float zs[CH_];
    int bt = blockIdx.x;
    int g  = threadIdx.x;                // 0..191
    if (g < CH_) zs[g] = zt[bt * CH_ + g];
    __syncthreads();
    float acc = bih[g];
    #pragma unroll
    for (int c = 0; c < CH_; ++c) acc += wih[g * CH_ + c] * zs[c];
    xg[bt * 192 + g] = acc;
}

// ---------------------------------------------------------------------------
// Phase 2b: sequential GRU over T + fused fc/sigmoid attention.
// One block per batch; recurrent weights register-resident (64 VGPR/thread).
// ---------------------------------------------------------------------------
__launch_bounds__(256, 1)
__global__ void gru_kernel(const float* __restrict__ xg,
                           const float* __restrict__ whh,
                           const float* __restrict__ bhh,
                           const float* __restrict__ fc_w,
                           const float* __restrict__ fc_b,
                           float* __restrict__ at_ws)
{
    __shared__ float hsh[GH_];
    __shared__ float hgb[3 * GH_];
    __shared__ float xgb[3 * GH_];
    const int b   = blockIdx.x;
    const int tid = threadIdx.x;

    float wreg[64];
    float bias = 0.f;
    if (tid < 192) {
        #pragma unroll
        for (int k = 0; k < 64; ++k) wreg[k] = whh[tid * 64 + k];
        bias = bhh[tid];
    } else if (tid < 224) {
        int o = tid - 192;
        #pragma unroll
        for (int k = 0; k < 64; ++k) wreg[k] = fc_w[o * 64 + k];
        bias = fc_b[o];
    } else {
        #pragma unroll
        for (int k = 0; k < 64; ++k) wreg[k] = 0.f;
    }
    if (tid < GH_) hsh[tid] = 0.f;
    __syncthreads();

    for (int t = 0; t < T_; ++t) {
        if (tid < 192) {
            float xv = xg[(b * T_ + t) * 192 + tid];   // in flight over matvec
            float acc = bias;
            #pragma unroll
            for (int k = 0; k < 64; ++k) acc += wreg[k] * hsh[k];
            hgb[tid] = acc;
            xgb[tid] = xv;
        }
        __syncthreads();
        if (tid < GH_) {
            float r = 1.f / (1.f + __expf(-(xgb[tid] + hgb[tid])));
            float z = 1.f / (1.f + __expf(-(xgb[64 + tid] + hgb[64 + tid])));
            float n = tanhf(xgb[128 + tid] + r * hgb[128 + tid]);
            hsh[tid] = (1.f - z) * n + z * hsh[tid];
        }
        __syncthreads();
        if (tid >= 192 && tid < 224) {
            float acc = bias;
            #pragma unroll
            for (int k = 0; k < 64; ++k) acc += wreg[k] * hsh[k];
            at_ws[(b * T_ + t) * CH_ + (tid - 192)] = 1.f / (1.f + __expf(-acc));
        }
    }
}

// ---------------------------------------------------------------------------
// Phase 3: out[b,2c]   = h * at,  out[b,2c+1] = x2   (float4 streams)
// ---------------------------------------------------------------------------
__global__ void final_kernel(const float* __restrict__ x,
                             const float* __restrict__ h_ws,
                             const float* __restrict__ at_ws,
                             float* __restrict__ out)
{
    int id   = blockIdx.x * blockDim.x + threadIdx.x;  // B*CH*T*(F/4)
    int f4   = id & 31;
    int rest = id >> 5;
    int t    = rest & (T_ - 1); rest >>= 9;
    int c    = rest & (CH_ - 1);
    int b    = rest >> 5;

    float a = at_ws[(b * T_ + t) * CH_ + c];
    const float4 h  = ((const float4*)h_ws)[((((b * CH_ + c) * T_ + t) * F_) >> 2) + f4];
    const float4 x2 = ((const float4*)x)[((((b * C_ + CH_ + c) * T_ + t) * F_) >> 2) + f4];
    float4 o1; o1.x = h.x * a; o1.y = h.y * a; o1.z = h.z * a; o1.w = h.w * a;
    ((float4*)out)[((((b * C_ + 2 * c) * T_ + t) * F_) >> 2) + f4]     = o1;
    ((float4*)out)[((((b * C_ + 2 * c + 1) * T_ + t) * F_) >> 2) + f4] = x2;
}

// ---------------------------------------------------------------------------
extern "C" void kernel_launch(void* const* d_in, const int* in_sizes, int n_in,
                              void* d_out, int out_size, void* d_ws, size_t ws_size,
                              hipStream_t stream)
{
    const float* x       = (const float*)d_in[0];
    const float* pw1_w   = (const float*)d_in[1];
    const float* pw1_b   = (const float*)d_in[2];
    const float* ln1_g   = (const float*)d_in[3];
    const float* ln1_b   = (const float*)d_in[4];
    const float* prelu1  = (const float*)d_in[5];
    const float* dw_w    = (const float*)d_in[6];
    const float* dw_b    = (const float*)d_in[7];
    const float* bn_g    = (const float*)d_in[8];
    const float* bn_b    = (const float*)d_in[9];
    const float* bn_mean = (const float*)d_in[10];
    const float* bn_var  = (const float*)d_in[11];
    const float* prelu2  = (const float*)d_in[12];
    const float* pw2_w   = (const float*)d_in[13];
    const float* pw2_b   = (const float*)d_in[14];
    const float* ln2_g   = (const float*)d_in[15];
    const float* ln2_b   = (const float*)d_in[16];
    const float* gru_wih = (const float*)d_in[17];
    const float* gru_whh = (const float*)d_in[18];
    const float* gru_bih = (const float*)d_in[19];
    const float* gru_bhh = (const float*)d_in[20];
    const float* fc_w    = (const float*)d_in[21];
    const float* fc_b    = (const float*)d_in[22];
    float* out = (float*)d_out;

    float* h_ws  = (float*)d_ws;                        // B*CH*T*F  f32
    float* zt_ws = h_ws  + (size_t)B_ * CH_ * T_ * F_;  // B*T*CH    f32
    float* xg_ws = zt_ws + (size_t)B_ * T_ * CH_;       // B*T*192   f32
    float* at_ws = xg_ws + (size_t)B_ * T_ * 192;       // B*T*CH    f32

    fused_conv_kernel<<<B_ * (T_ / TILE_T), 256, 0, stream>>>(
        x, pw1_w, pw1_b, ln1_g, ln1_b, prelu1, dw_w, dw_b,
        bn_g, bn_b, bn_mean, bn_var, prelu2, pw2_w, pw2_b,
        ln2_g, ln2_b, h_ws, zt_ws);

    xg_kernel<<<B_ * T_, 192, 0, stream>>>(zt_ws, gru_wih, gru_bih, xg_ws);

    gru_kernel<<<B_, 256, 0, stream>>>(xg_ws, gru_whh, gru_bhh, fc_w, fc_b, at_ws);

    final_kernel<<<(B_ * CH_ * T_ * (F_ / 4)) / 256, 256, 0, stream>>>(
        x, h_ws, at_ws, out);
}
